// QuestionAnswering_67559835566517
// MI455X (gfx1250) — matrix operand order repaired and checked by reading, not compiled, not run on hardware
//
#include <hip/hip_runtime.h>
#include <math.h>

#define DEV __device__ __forceinline__

static constexpr int kH   = 768;
static constexpr int kNH  = 12;
static constexpr int kDH  = 64;
static constexpr int kSeg = 512;
static constexpr int kB   = 8;
static constexpr int kS   = 4096;
static constexpr int kL   = 2;

typedef __bf16 v16bf __attribute__((ext_vector_type(16)));
typedef float  v8f   __attribute__((ext_vector_type(8)));
typedef unsigned int v4u __attribute__((ext_vector_type(4)));
typedef unsigned short u16;

DEV u16 f2bf(float f) {
  unsigned u = __float_as_uint(f);
  u += 0x7FFFu + ((u >> 16) & 1u);           // round-to-nearest-even
  return (u16)(u >> 16);
}
DEV float bf2f(u16 h) { return __uint_as_float(((unsigned)h) << 16); }
DEV float gelu_t(float x) {
  return 0.5f * x * (1.f + tanhf(0.7978845608028654f * (x + 0.044715f * x * x * x)));
}

DEV float block_sum(float v) {
  __shared__ float sb[256];
  int t = threadIdx.x; sb[t] = v; __syncthreads();
  for (int s = 128; s > 0; s >>= 1) { if (t < s) sb[t] += sb[t + s]; __syncthreads(); }
  float r = sb[0]; __syncthreads(); return r;
}
DEV float block_max(float v) {
  __shared__ float sm[256];
  int t = threadIdx.x; sm[t] = v; __syncthreads();
  for (int s = 128; s > 0; s >>= 1) { if (t < s) sm[t] = fmaxf(sm[t], sm[t + s]); __syncthreads(); }
  float r = sm[0]; __syncthreads(); return r;
}

// ---------------------------------------------------------------------------
// Generic bf16 WMMA GEMM: C[z] = A[z] (MxK) * B[z] (KxN) (+epilogue)
//   A element (m,k)  at A [aBase + m*lda + k]              (bf16, lda mult of 8)
//   B element (k,n)  at Bm[bBase + k*ldbK + n*ldbN]        (bf16)
//   C element (m,n)  at    cBase + m*ldc + n               (f32 and/or bf16 out)
//   batch base = (z/zdiv)*Outer + (z%zdiv)*Inner
// EPI: 0=none 1=+bias[n] 2=gelu(x+bias[n]) 3=x*scale + (1-mask[(z/zdiv)*maskN+n])*-1e4
// BMODE: 0 -> ldbN==1 (N-contiguous: b128 load + scalar transposed LDS store)
//        1 -> ldbK==1 (K-contiguous: b128 load + b128 LDS store, no transpose)
// Tile: 128x64, K-step 32, 8 waves x (16x64) per workgroup, wave32.
// ---------------------------------------------------------------------------
template <int EPI, int BMODE>
__global__ __launch_bounds__(256)
void gemm_bf16_kernel(const u16* __restrict__ A, int lda, long long aO, long long aI,
                      const u16* __restrict__ Bm, int ldbK, int ldbN, long long bO, long long bI,
                      const float* __restrict__ bias,
                      float* __restrict__ Cf, u16* __restrict__ Cbf,
                      int ldc, long long cO, long long cI,
                      int N, int K, int zdiv, float scale,
                      const float* __restrict__ mask, int maskN) {
  __shared__ alignas(16) u16 lsA[128 * 40];
  __shared__ alignas(16) u16 lsB[64 * 40];
  const int tid = threadIdx.x, lane = tid & 31, wave = tid >> 5;
  const int z = blockIdx.z, zo = z / zdiv, zi = z % zdiv;
  const long long Ab = (long long)zo * aO + (long long)zi * aI;
  const long long Bb = (long long)zo * bO + (long long)zi * bI;
  const long long Cb = (long long)zo * cO + (long long)zi * cI;
  const int m0 = blockIdx.y * 128, n0 = blockIdx.x * 64;

  v8f acc[4];
  v8f vz = {0.f, 0.f, 0.f, 0.f, 0.f, 0.f, 0.f, 0.f};
#pragma unroll
  for (int i = 0; i < 4; i++) acc[i] = vz;

  const int arow = tid >> 1, ahalf = tid & 1;       // A tile: 128 rows x 32, 16 elems/thread
  const int bk = tid >> 3, bn = (tid & 7) * 8;      // BMODE 0: 32 k x (8 threads x 8 n)
  const int bn1 = tid >> 2, bkc = (tid & 3) * 8;    // BMODE 1: 64 n x (4 threads x 8 k)
  const int ktiles = K >> 5;

  for (int kt = 0; kt < ktiles; ++kt) {
    const int k0 = kt << 5;
    {   // stage A tile (row-major, pad 40)
      const u16* src = A + Ab + (long long)(m0 + arow) * lda + k0 + ahalf * 16;
      const v4u* p = (const v4u*)src;
      v4u x0 = p[0], x1 = p[1];
      v4u* q = (v4u*)&lsA[arow * 40 + ahalf * 16];
      q[0] = x0; q[1] = x1;
      if (kt + 1 < ktiles) __builtin_prefetch(src + 32, 0, 3);
    }
    if (BMODE == 0) {   // B N-contiguous: b128 load, scalar transposed stores
      const long long rb = Bb + (long long)(k0 + bk) * ldbK;
      union { v4u v; u16 h[8]; } bld;
      bld.v = *(const v4u*)(Bm + rb + (n0 + bn));
#pragma unroll
      for (int j = 0; j < 8; j++) lsB[(bn + j) * 40 + bk] = bld.h[j];
    } else {            // B K-contiguous: b128 load + b128 store (lsB[n][k] is K-major)
      const v4u bv = *(const v4u*)(Bm + Bb + (long long)(n0 + bn1) * ldbN + k0 + bkc);
      *(v4u*)&lsB[bn1 * 40 + bkc] = bv;
    }
    __syncthreads();

    union Fr { v16bf v; v4u q[2]; };
    Fr fa;
    {   // A fragment per ISA 16-bit 16x32 layout
      const int mrow = (wave << 4) + (lane & 15);
      const int ksel = (lane < 16) ? 0 : 8;
      fa.q[0] = *(const v4u*)&lsA[mrow * 40 + ksel];        // K {0..7} or {8..15}
      fa.q[1] = *(const v4u*)&lsA[mrow * 40 + 16 + ksel];   // K {16..23} or {24..31}
    }
    const int kselB = (lane < 16) ? 0 : 8;                  // B: K {0..7,16..23} / {8..15,24..31}
#pragma unroll
    for (int nf = 0; nf < 4; ++nf) {
      Fr fb;
      const int ncol = (nf << 4) + (lane & 15);
      fb.q[0] = *(const v4u*)&lsB[ncol * 40 + kselB];
      fb.q[1] = *(const v4u*)&lsB[ncol * 40 + 16 + kselB];
      acc[nf] = __builtin_amdgcn_wmma_f32_16x16x32_bf16(
          false, fa.v, false, fb.v, (short)0, acc[nf], false, false);
    }
    __syncthreads();
  }

  // epilogue: C 16x16 f32 layout -> VGPR i = row i (lanes 0-15) / row 8+i (lanes 16-31)
  const int rbase = m0 + (wave << 4) + ((lane < 16) ? 0 : 8);
  const int nlane = n0 + (lane & 15);
#pragma unroll
  for (int nf = 0; nf < 4; ++nf) {
    const int c = nlane + nf * 16;
    float badd = 0.f, madd = 0.f;
    if (EPI == 1 || EPI == 2) badd = bias[c];
    if (EPI == 3) madd = (1.f - mask[(long long)zo * maskN + c]) * -10000.f;
#pragma unroll
    for (int i = 0; i < 8; i++) {
      float v = acc[nf][i];
      if (EPI == 1) v += badd;
      else if (EPI == 2) v = gelu_t(v + badd);
      else if (EPI == 3) v = v * scale + madd;
      const long long idx = Cb + (long long)(rbase + i) * ldc + c;
      if (Cf)  Cf[idx] = v;
      if (Cbf) Cbf[idx] = f2bf(v);
    }
  }
}

// --------------------------- elementwise / reduction kernels ----------------

__global__ void f32_to_bf16_kernel(const float* __restrict__ in, u16* __restrict__ out,
                                   long long n) {
  long long i = (long long)blockIdx.x * 256 + threadIdx.x;
  const long long st = (long long)gridDim.x * 256;
  for (; i < n; i += st) out[i] = f2bf(in[i]);
}

__global__ __launch_bounds__(256)
void pool_scatter_kernel(const float* __restrict__ seq, const int* __restrict__ ids,
                         float* __restrict__ pool) {
  const int bs = blockIdx.x;                 // b*kS + s
  const int b = bs / kS;
  const int id = ids[bs];
  const float* s = seq + (long long)bs * kH;
  float* d = pool + ((long long)b * kSeg + id) * kH;
  for (int h = threadIdx.x; h < kH; h += 256) atomicAdd(&d[h], s[h]);
}

__global__ __launch_bounds__(256)
void add_ln_kernel(float* __restrict__ x, const float* __restrict__ dl,
                   const float* __restrict__ g, const float* __restrict__ bta,
                   u16* __restrict__ xbf) {
  const long long base = (long long)blockIdx.x * kH;
  float vals[3]; float s = 0.f, sq = 0.f;
#pragma unroll
  for (int j = 0; j < 3; j++) {
    const int i = threadIdx.x + j * 256;
    const float v = x[base + i] + dl[base + i];
    vals[j] = v; s += v; sq += v * v;
  }
  s = block_sum(s); sq = block_sum(sq);
  const float mean = s * (1.f / kH);
  const float var = sq * (1.f / kH) - mean * mean;
  const float rstd = rsqrtf(var + 1e-12f);
#pragma unroll
  for (int j = 0; j < 3; j++) {
    const int i = threadIdx.x + j * 256;
    const float o = (vals[j] - mean) * rstd * g[i] + bta[i];
    x[base + i] = o; xbf[base + i] = f2bf(o);
  }
}

__global__ __launch_bounds__(256)
void softmax512_kernel(const float* __restrict__ sc, u16* __restrict__ pr) {
  const long long base = (long long)blockIdx.x * kSeg;
  const float v0 = sc[base + threadIdx.x], v1 = sc[base + threadIdx.x + 256];
  const float m = block_max(fmaxf(v0, v1));
  const float e0 = expf(v0 - m), e1 = expf(v1 - m);
  const float inv = 1.f / block_sum(e0 + e1);
  pr[base + threadIdx.x]       = f2bf(e0 * inv);
  pr[base + threadIdx.x + 256] = f2bf(e1 * inv);
}

__global__ __launch_bounds__(256)
void build_cat_kernel(const float* __restrict__ seq, const float* __restrict__ rh,
                      const float* __restrict__ ch, const int* __restrict__ rid,
                      const int* __restrict__ cid, u16* __restrict__ cat) {
  const int bs = blockIdx.x; const int b = bs / kS;
  const float* s = seq + (long long)bs * kH;
  const float* r = rh + ((long long)b * kSeg + rid[bs]) * kH;
  const float* c = ch + ((long long)b * kSeg + cid[bs]) * kH;
  u16* o = cat + (long long)bs * (3 * kH);
  for (int i = threadIdx.x; i < kH; i += 256) {
    o[i] = f2bf(s[i]); o[kH + i] = f2bf(r[i]); o[2 * kH + i] = f2bf(c[i]);
  }
}

template <bool BFIN>
__global__ __launch_bounds__(256)
void matvec2_kernel(const void* __restrict__ Ain, const float* __restrict__ w,
                    const float* __restrict__ bias, float* __restrict__ out, int Kd) {
  const long long row = blockIdx.x;
  float a0 = 0.f, a1 = 0.f;
  for (int k = threadIdx.x; k < Kd; k += 256) {
    const float av = BFIN ? bf2f(((const u16*)Ain)[row * Kd + k])
                          : ((const float*)Ain)[row * Kd + k];
    a0 += av * w[k * 2]; a1 += av * w[k * 2 + 1];
  }
  a0 = block_sum(a0); a1 = block_sum(a1);
  if (threadIdx.x == 0) { out[row * 2] = a0 + bias[0]; out[row * 2 + 1] = a1 + bias[1]; }
}

__global__ __launch_bounds__(256)
void blend_kernel(const float* __restrict__ l1, const float* __restrict__ l2,
                  const float* __restrict__ wts, float* __restrict__ out) {
  const int b = blockIdx.x, side = blockIdx.y;
  const long long base = (long long)b * kS;
  float m1l = -1e30f, m2l = -1e30f;
  for (int s = threadIdx.x; s < kS; s += 256) {
    m1l = fmaxf(m1l, l1[(base + s) * 2 + side]);
    m2l = fmaxf(m2l, l2[(base + s) * 2 + side]);
  }
  const float m1 = block_max(m1l), m2 = block_max(m2l);
  float s1l = 0.f, s2l = 0.f;
  for (int s = threadIdx.x; s < kS; s += 256) {
    s1l += expf(l1[(base + s) * 2 + side] - m1);
    s2l += expf(l2[(base + s) * 2 + side] - m2);
  }
  const float i1 = 1.f / block_sum(s1l), i2 = 1.f / block_sum(s2l);
  for (int s = threadIdx.x; s < kS; s += 256) {
    const float w = wts[base + s];
    const float p1 = expf(l1[(base + s) * 2 + side] - m1) * i1;
    const float p2 = expf(l2[(base + s) * 2 + side] - m2) * i2;
    out[(long long)side * kB * kS + base + s] = p1 * (1.f - w) + p2 * w;
  }
}

// ------------------------------- host side ----------------------------------
// Only 4 template combos are instantiated (the ones actually used):
//   epi 0/1/2 with BMODE=0 (ldbN==1 operands), epi 3 with BMODE=1 (scores, ldbK==1)

static void run_gemm(hipStream_t st, int epi,
                     const u16* A, int lda, long long aO, long long aI,
                     const u16* Bm, int ldbK, int ldbN, long long bO, long long bI,
                     const float* bias, float* Cf, u16* Cbf,
                     int ldc, long long cO, long long cI,
                     int M, int N, int K, int batches, int zdiv,
                     float scale, const float* mask, int maskN) {
  dim3 g(N / 64, M / 128, batches), b(256, 1, 1);
  switch (epi) {
    case 0: gemm_bf16_kernel<0, 0><<<g, b, 0, st>>>(A, lda, aO, aI, Bm, ldbK, ldbN, bO, bI,
              bias, Cf, Cbf, ldc, cO, cI, N, K, zdiv, scale, mask, maskN); break;
    case 1: gemm_bf16_kernel<1, 0><<<g, b, 0, st>>>(A, lda, aO, aI, Bm, ldbK, ldbN, bO, bI,
              bias, Cf, Cbf, ldc, cO, cI, N, K, zdiv, scale, mask, maskN); break;
    case 2: gemm_bf16_kernel<2, 0><<<g, b, 0, st>>>(A, lda, aO, aI, Bm, ldbK, ldbN, bO, bI,
              bias, Cf, Cbf, ldc, cO, cI, N, K, zdiv, scale, mask, maskN); break;
    default: gemm_bf16_kernel<3, 1><<<g, b, 0, st>>>(A, lda, aO, aI, Bm, ldbK, ldbN, bO, bI,
              bias, Cf, Cbf, ldc, cO, cI, N, K, zdiv, scale, mask, maskN); break;
  }
}

static void to_bf16(hipStream_t st, const float* s, u16* d, long long n) {
  long long nb = (n + 255) / 256; if (nb > 4096) nb = 4096;
  f32_to_bf16_kernel<<<dim3((unsigned)nb), 256, 0, st>>>(s, d, n);
}

static void run_encoder(hipStream_t st, float* x_f, u16* x_bf,
                        const u16* aw, const float* ab,
                        const u16* w1, const float* b1,
                        const u16* w2, const float* b2,
                        const float* lg, const float* lb, const float* mask,
                        u16* q, u16* k, u16* v, u16* ctx,
                        float* scores, u16* probs, float* tmp, u16* ffh) {
  const int M = kB * kSeg;                       // 4096
  const long long HH = (long long)kH * kH;
  const long long SH = (long long)kSeg * kH;     // per-batch activation stride
  const long long SS = (long long)kSeg * kSeg;   // per-head score stride
  for (int l = 0; l < kL; l++) {
    const u16* Wq = aw + (long long)(l * 4 + 0) * HH;
    const u16* Wk = aw + (long long)(l * 4 + 1) * HH;
    const u16* Wv = aw + (long long)(l * 4 + 2) * HH;
    const u16* Wo = aw + (long long)(l * 4 + 3) * HH;
    run_gemm(st, 1, x_bf, kH, 0, 0, Wq, kH, 1, 0, 0, ab + (l * 4 + 0) * kH,
             nullptr, q, kH, 0, 0, M, kH, kH, 1, 1, 0.f, nullptr, 0);
    run_gemm(st, 1, x_bf, kH, 0, 0, Wk, kH, 1, 0, 0, ab + (l * 4 + 1) * kH,
             nullptr, k, kH, 0, 0, M, kH, kH, 1, 1, 0.f, nullptr, 0);
    run_gemm(st, 1, x_bf, kH, 0, 0, Wv, kH, 1, 0, 0, ab + (l * 4 + 2) * kH,
             nullptr, v, kH, 0, 0, M, kH, kH, 1, 1, 0.f, nullptr, 0);
    // scores[b,h,q,kk] = (Q . K^T) * 1/8 + additive mask ; batched over b*NH
    run_gemm(st, 3, q, kH, SH, kDH,
             k, 1, kH, SH, kDH,
             nullptr, scores, nullptr, kSeg, (long long)kNH * SS, SS,
             kSeg, kSeg, kDH, kB * kNH, kNH, 0.125f, mask, kSeg);
    softmax512_kernel<<<kB * kNH * kSeg, 256, 0, st>>>(scores, probs);
    // ctx[b,q,h,:] = P @ V
    run_gemm(st, 0, probs, kSeg, (long long)kNH * SS, SS,
             v, kH, 1, SH, kDH,
             nullptr, nullptr, ctx, kH, SH, kDH,
             kSeg, kDH, kSeg, kB * kNH, kNH, 0.f, nullptr, 0);
    run_gemm(st, 1, ctx, kH, 0, 0, Wo, kH, 1, 0, 0, ab + (l * 4 + 3) * kH,
             tmp, nullptr, kH, 0, 0, M, kH, kH, 1, 1, 0.f, nullptr, 0);
    add_ln_kernel<<<M, 256, 0, st>>>(x_f, tmp, lg + (l * 2 + 0) * kH, lb + (l * 2 + 0) * kH, x_bf);
    run_gemm(st, 2, x_bf, kH, 0, 0, w1 + (long long)l * kH * 4 * kH, 4 * kH, 1, 0, 0,
             b1 + (long long)l * 4 * kH, nullptr, ffh, 4 * kH, 0, 0,
             M, 4 * kH, kH, 1, 1, 0.f, nullptr, 0);
    run_gemm(st, 1, ffh, 4 * kH, 0, 0, w2 + (long long)l * 4 * kH * kH, kH, 1, 0, 0,
             b2 + (long long)l * kH, tmp, nullptr, kH, 0, 0,
             M, kH, 4 * kH, 1, 1, 0.f, nullptr, 0);
    add_ln_kernel<<<M, 256, 0, st>>>(x_f, tmp, lg + (l * 2 + 1) * kH, lb + (l * 2 + 1) * kH, x_bf);
  }
}

extern "C" void kernel_launch(void* const* d_in, const int* in_sizes, int n_in,
                              void* d_out, int out_size, void* d_ws, size_t ws_size,
                              hipStream_t stream) {
  (void)in_sizes; (void)n_in; (void)out_size;
  const float* seq  = (const float*)d_in[0];
  const float* mrow = (const float*)d_in[1];
  const float* mcol = (const float*)d_in[2];
  const float* iw   = (const float*)d_in[3];
  const int*   rid  = (const int*)d_in[4];
  const int*   cid  = (const int*)d_in[5];
  const float* rhw  = (const float*)d_in[6];
  const float* rhb  = (const float*)d_in[7];
  const float* chw  = (const float*)d_in[8];
  const float* chb  = (const float*)d_in[9];
  const float* r_aw = (const float*)d_in[10];
  const float* r_ab = (const float*)d_in[11];
  const float* r_f1 = (const float*)d_in[12];
  const float* r_b1 = (const float*)d_in[13];
  const float* r_f2 = (const float*)d_in[14];
  const float* r_b2 = (const float*)d_in[15];
  const float* r_lg = (const float*)d_in[16];
  const float* r_lb = (const float*)d_in[17];
  const float* c_aw = (const float*)d_in[18];
  const float* c_ab = (const float*)d_in[19];
  const float* c_f1 = (const float*)d_in[20];
  const float* c_b1 = (const float*)d_in[21];
  const float* c_f2 = (const float*)d_in[22];
  const float* c_b2 = (const float*)d_in[23];
  const float* c_lg = (const float*)d_in[24];
  const float* c_lb = (const float*)d_in[25];
  const float* qaw  = (const float*)d_in[26];
  const float* qab  = (const float*)d_in[27];
  const float* h1w  = (const float*)d_in[28];
  const float* h1b  = (const float*)d_in[29];
  const float* h2w  = (const float*)d_in[30];
  const float* h2b  = (const float*)d_in[31];
  const float* tw   = (const float*)d_in[32];
  const float* tb   = (const float*)d_in[33];

  char* wsb = (char*)d_ws; size_t off = 0;
  auto carve = [&](long long bytes) -> void* {
    off = (off + 255) & ~(size_t)255; void* p = wsb + off; off += (size_t)bytes; return p;
  };
  const long long HH = (long long)kH * kH;
  const long long MP = (long long)kB * kSeg;   // 4096 encoder rows
  const long long MT = (long long)kB * kS;     // 32768 token rows

  u16* rhw_bf = (u16*)carve(HH * 2);
  u16* chw_bf = (u16*)carve(HH * 2);
  u16* raw_bf = (u16*)carve((long long)kL * 4 * HH * 2);
  u16* rf1_bf = (u16*)carve((long long)kL * kH * 4 * kH * 2);
  u16* rf2_bf = (u16*)carve((long long)kL * 4 * kH * kH * 2);
  u16* caw_bf = (u16*)carve((long long)kL * 4 * HH * 2);
  u16* cf1_bf = (u16*)carve((long long)kL * kH * 4 * kH * 2);
  u16* cf2_bf = (u16*)carve((long long)kL * 4 * kH * kH * 2);
  u16* h1w_bf = (u16*)carve((long long)3 * kH * 2 * kH * 2);
  u16* h2w_bf = (u16*)carve((long long)2 * kH * kH * 2);

  float* pool_f = (float*)carve(MP * kH * 4);
  u16*   pool_b = (u16*)carve(MP * kH * 2);
  float* rx_f = (float*)carve(MP * kH * 4);  u16* rx_bf = (u16*)carve(MP * kH * 2);
  float* cx_f = (float*)carve(MP * kH * 4);  u16* cx_bf = (u16*)carve(MP * kH * 2);
  u16* q_bf   = (u16*)carve(MP * kH * 2);
  u16* k_bf   = (u16*)carve(MP * kH * 2);
  u16* v_bf   = (u16*)carve(MP * kH * 2);
  u16* ctx_bf = (u16*)carve(MP * kH * 2);
  float* scores = (float*)carve((long long)kB * kNH * kSeg * kSeg * 4);
  u16*   probs  = (u16*)carve((long long)kB * kNH * kSeg * kSeg * 2);
  float* tmp_f  = (float*)carve(MP * kH * 4);
  u16*   ffh_bf = (u16*)carve(MP * 4 * kH * 2);
  u16* cat_bf = (u16*)carve(MT * 3 * kH * 2);
  u16* h1_bf  = (u16*)carve(MT * 2 * kH * 2);
  u16* h2_bf  = (u16*)carve(MT * kH * 2);
  float* lg1  = (float*)carve(MT * 2 * 4);
  float* lg2  = (float*)carve(MT * 2 * 4);
  if (off > ws_size) return;   // insufficient scratch; bail deterministically

  // weights -> bf16 (deterministic, redone every call)
  to_bf16(stream, rhw, rhw_bf, HH);
  to_bf16(stream, chw, chw_bf, HH);
  to_bf16(stream, r_aw, raw_bf, (long long)kL * 4 * HH);
  to_bf16(stream, r_f1, rf1_bf, (long long)kL * kH * 4 * kH);
  to_bf16(stream, r_f2, rf2_bf, (long long)kL * 4 * kH * kH);
  to_bf16(stream, c_aw, caw_bf, (long long)kL * 4 * HH);
  to_bf16(stream, c_f1, cf1_bf, (long long)kL * kH * 4 * kH);
  to_bf16(stream, c_f2, cf2_bf, (long long)kL * 4 * kH * kH);
  to_bf16(stream, h1w, h1w_bf, (long long)3 * kH * 2 * kH);
  to_bf16(stream, h2w, h2w_bf, (long long)2 * kH * kH);

  // ----- row branch: segment-sum pool -> gelu(dense) -> 2-layer encoder -----
  hipMemsetAsync(pool_f, 0, (size_t)(MP * kH * 4), stream);
  pool_scatter_kernel<<<kB * kS, 256, 0, stream>>>(seq, rid, pool_f);
  to_bf16(stream, pool_f, pool_b, MP * kH);
  run_gemm(stream, 2, pool_b, kH, 0, 0, rhw_bf, kH, 1, 0, 0, rhb,
           rx_f, rx_bf, kH, 0, 0, (int)MP, kH, kH, 1, 1, 0.f, nullptr, 0);
  run_encoder(stream, rx_f, rx_bf, raw_bf, r_ab, rf1_bf, r_b1, rf2_bf, r_b2,
              r_lg, r_lb, mrow, q_bf, k_bf, v_bf, ctx_bf, scores, probs, tmp_f, ffh_bf);

  // ----- col branch -----
  hipMemsetAsync(pool_f, 0, (size_t)(MP * kH * 4), stream);
  pool_scatter_kernel<<<kB * kS, 256, 0, stream>>>(seq, cid, pool_f);
  to_bf16(stream, pool_f, pool_b, MP * kH);
  run_gemm(stream, 2, pool_b, kH, 0, 0, chw_bf, kH, 1, 0, 0, chb,
           cx_f, cx_bf, kH, 0, 0, (int)MP, kH, kH, 1, 1, 0.f, nullptr, 0);
  run_encoder(stream, cx_f, cx_bf, caw_bf, c_ab, cf1_bf, c_b1, cf2_bf, c_b2,
              c_lg, c_lb, mcol, q_bf, k_bf, v_bf, ctx_bf, scores, probs, tmp_f, ffh_bf);

  // ----- QA head -----
  matvec2_kernel<false><<<(int)MT, 256, 0, stream>>>(seq, qaw, qab, lg1, kH);
  build_cat_kernel<<<(int)MT, 256, 0, stream>>>(seq, rx_f, cx_f, rid, cid, cat_bf);
  run_gemm(stream, 2, cat_bf, 3 * kH, 0, 0, h1w_bf, 2 * kH, 1, 0, 0, h1b,
           nullptr, h1_bf, 2 * kH, 0, 0, (int)MT, 2 * kH, 3 * kH, 1, 1, 0.f, nullptr, 0);
  run_gemm(stream, 2, h1_bf, 2 * kH, 0, 0, h2w_bf, kH, 1, 0, 0, h2b,
           nullptr, h2_bf, kH, 0, 0, (int)MT, kH, 2 * kH, 1, 1, 0.f, nullptr, 0);
  matvec2_kernel<true><<<(int)MT, 256, 0, stream>>>(h2_bf, tw, tb, lg2, kH);
  blend_kernel<<<dim3(kB, 2), 256, 0, stream>>>(lg1, lg2, iw, (float*)d_out);
}
